// TraditionalSP_76776835383940
// MI455X (gfx1250) — matrix operand order repaired and checked by reading, not compiled, not run on hardware
//
#include <hip/hip_runtime.h>
#include <stdint.h>
#include <stddef.h>

// ---------------------------------------------------------------------------
// HPS piano-key quantile-threshold kernel for MI455X (gfx1250).
//  - key gather + harmonic product (in log space) == 88x128 GEMM -> WMMA f32
//    (A-matrix synthesized in registers from key_bins; one A-fragment feeds
//     4 WMMAs via 4 time-tile accumulators -> WMMA-dense inner loop)
//  - mel tile staged via Tensor Data Mover (tensor_load_to_lds)
//  - probs tile written back via tensor_store_from_lds (x2 outputs)
//  - quantile threshold reduced to rank count: out = (#{v_j > v_k} <= 13)
// ---------------------------------------------------------------------------

typedef float    v2f   __attribute__((ext_vector_type(2)));
typedef float    v8f   __attribute__((ext_vector_type(8)));
typedef _Float16 v16h  __attribute__((ext_vector_type(16)));
typedef uint32_t u32x4 __attribute__((ext_vector_type(4)));
typedef int      i32x4 __attribute__((ext_vector_type(4)));
typedef int      i32x8 __attribute__((ext_vector_type(8)));

#define MEL_BINS 128
#define NUM_KEYS 88
#define KPAD     96           // keys padded to 6 tiles of 16
#define TT       64           // time-tile per workgroup
#define T_TOTAL  8192
#define NBATCH   32
#define NTHREADS 256          // 8 waves (wave32)

#if defined(__HIP_DEVICE_COMPILE__) && \
    __has_builtin(__builtin_amdgcn_tensor_load_to_lds) && \
    __has_builtin(__builtin_amdgcn_tensor_store_from_lds) && \
    __has_builtin(__builtin_amdgcn_s_wait_tensorcnt)
#define HAVE_TDM 1
#else
#define HAVE_TDM 0
#endif

#if defined(__HIP_DEVICE_COMPILE__) && \
    __has_builtin(__builtin_amdgcn_wmma_f32_16x16x4_f32)
#define HAVE_WMMA_F32X4 1
#else
#define HAVE_WMMA_F32X4 0
#endif

#if !HAVE_WMMA_F32X4 && defined(__HIP_DEVICE_COMPILE__) && \
    __has_builtin(__builtin_amdgcn_wmma_f32_16x16x32_f16)
#define HAVE_WMMA_F16 1
#else
#define HAVE_WMMA_F16 0
#endif

// Low 32 bits of a flat pointer into LDS == LDS byte offset (ISA: LDS aperture
// maps addr[31:0] directly onto LDS).
__device__ __forceinline__ uint32_t lds_off(const void* p) {
  return (uint32_t)(uintptr_t)p;
}

#if HAVE_TDM
// Build a 2D-tile Tensor DMA Descriptor (D#) per CDNA5 ISA ch.8 and issue a
// TDM load or store. data_size = 4 bytes. Groups 2/3 unused (<=2D tensor).
// clang-23 builtin arity: (u32x4 g0, i32x8 g1, i32x4, i32x4, i32x8, i32 cpol).
__device__ __forceinline__ void tdm_copy_2d(uint32_t lds_addr, const void* gptr,
                                            uint32_t tensor_d0, uint32_t tensor_d1,
                                            uint32_t tile_d0,  uint32_t tile_d1,
                                            uint64_t stride0_elems, bool to_lds)
{
  const uint64_t ga = (uint64_t)(uintptr_t)gptr;
  u32x4 g0;
  g0.x = 1u;                                              // count=1, user D#
  g0.y = lds_addr;                                        // [63:32] lds_addr
  g0.z = (uint32_t)ga;                                    // [95:64] global lo
  g0.w = (uint32_t)((ga >> 32) & 0x01FFFFFFu)             // [120:96] global hi
       | 0x80000000u;                                     // [127:126] type=2
  i32x8 g1;
  g1[0] = (int)(2u << 16);                                // data_size=2 -> 4B
  g1[1] = (int)((tensor_d0 & 0xFFFFu) << 16);             // tensor_dim0 lo16
  g1[2] = (int)(((tensor_d0 >> 16) & 0xFFFFu)             // tensor_dim0 hi16
              | ((tensor_d1 & 0xFFFFu) << 16));           // tensor_dim1 lo16
  g1[3] = (int)(((tensor_d1 >> 16) & 0xFFFFu)             // tensor_dim1 hi16
              | ((tile_d0 & 0xFFFFu) << 16));             // tile_dim0
  g1[4] = (int)(tile_d1 & 0xFFFFu);                       // tile_dim1 (dim2=0)
  g1[5] = (int)(uint32_t)(stride0_elems & 0xFFFFFFFFull); // dim0_stride lo32
  g1[6] = (int)(uint32_t)((stride0_elems >> 32) & 0xFFFFull); // dim0_stride hi16
  g1[7] = 0;                                              // dim1_stride (unused)
  i32x4 z4 = {0, 0, 0, 0};
  i32x8 z8 = {0, 0, 0, 0, 0, 0, 0, 0};
  if (to_lds) __builtin_amdgcn_tensor_load_to_lds(g0, g1, z4, z4, z8, 0);
  else        __builtin_amdgcn_tensor_store_from_lds(g0, g1, z4, z4, z8, 0);
}
#endif

// A[key][k] of the 88x128 selection/harmonic matrix, synthesized on the fly.
// bk < 0 => padded key row (all zero).
__device__ __forceinline__ float a_entry(int bk, int k) {
  float v = 0.0f;
  if (k == bk)                 v += 1.0f;
  if (bk < 64 && k == 2 * bk)  v += 1.0f;   // h=2 harmonic (row < 64)
  if (bk < 43 && k == 3 * bk)  v += 1.0f;   // h=3 harmonic (row < 43)
  return v;
}

__global__ __launch_bounds__(NTHREADS)
void hps_key_quantile_kernel(const float* __restrict__ mel,
                             const int*   __restrict__ key_bins,
                             float*       __restrict__ out)
{
  __shared__ float Ml[MEL_BINS * TT];   // mel tile  [bin][t]   (32 KB)
  __shared__ float Vl[KPAD * TT];       // log-energy [key][t]; reused as probs
  __shared__ int   kb[KPAD];

  const int tid = threadIdx.x;
  const int b   = blockIdx.y;
  const int t0  = blockIdx.x * TT;
  const float* mel_tile = mel + ((size_t)b * MEL_BINS) * T_TOTAL + t0;

  // ---- Stage mel tile (128 rows x 64 contiguous floats, row stride 8192) ----
#if HAVE_TDM
  if (tid < 32)
    tdm_copy_2d(lds_off(Ml), mel_tile, T_TOTAL, MEL_BINS, TT, MEL_BINS,
                (uint64_t)T_TOTAL, /*to_lds=*/true);
#else
  for (int i = tid; i < MEL_BINS * TT; i += NTHREADS)
    Ml[i] = mel_tile[(size_t)(i / TT) * T_TOTAL + (i % TT)];
#endif
  if (tid < KPAD) kb[tid] = (tid < NUM_KEYS) ? key_bins[tid] : -1;
#if HAVE_TDM
  if (tid < 32) __builtin_amdgcn_s_wait_tensorcnt(0);
#endif
  __syncthreads();

  // ---- GEMM: V(96xTT) = A(96x128) * Ml(128xTT) via WMMA 16x16 tiles --------
  const int wave = tid >> 5;
  const int lane = tid & 31;
  const int half = lane >> 4;   // 0: lanes 0-15, 1: lanes 16-31
  const int mrow = lane & 15;

#if HAVE_WMMA_F32X4
  // Wave w (w<6) owns key-tile kt=w and all 4 time sub-tiles: one register-
  // synthesized A fragment feeds 4 back-to-back WMMAs (4 accumulators).
  if (wave < 6) {
    const int kt = wave;
    const int bk = kb[kt * 16 + mrow];            // this lane's A-matrix row
    const float* Bbase = Ml + mrow;               // + nt*16 selects sub-tile
    v8f acc0 = {0.f,0.f,0.f,0.f,0.f,0.f,0.f,0.f};
    v8f acc1 = acc0, acc2 = acc0, acc3 = acc0;
#pragma unroll 2
    for (int k0 = 0; k0 < MEL_BINS; k0 += 4) {
      const int ka = k0 + 2 * half;               // A 16x4 / B 4x16 layouts
      v2f a;
      a.x = a_entry(bk, ka);
      a.y = a_entry(bk, ka + 1);
      const float* r0 = Bbase + ka * TT;          // row ka
      const float* r1 = Bbase + (ka + 1) * TT;    // row ka+1
      v2f b0, b1, b2, b3;
      b0.x = r0[ 0]; b0.y = r1[ 0];
      b1.x = r0[16]; b1.y = r1[16];
      b2.x = r0[32]; b2.y = r1[32];
      b3.x = r0[48]; b3.y = r1[48];
      acc0 = __builtin_amdgcn_wmma_f32_16x16x4_f32(false, a, false, b0,
                                                   (short)0, acc0, false, false);
      acc1 = __builtin_amdgcn_wmma_f32_16x16x4_f32(false, a, false, b1,
                                                   (short)0, acc1, false, false);
      acc2 = __builtin_amdgcn_wmma_f32_16x16x4_f32(false, a, false, b2,
                                                   (short)0, acc2, false, false);
      acc3 = __builtin_amdgcn_wmma_f32_16x16x4_f32(false, a, false, b3,
                                                   (short)0, acc3, false, false);
    }
    // D layout: VGPR i holds M=i (lanes 0-15) / M=i+8 (lanes 16-31), N=lane%16
    float* vrow = Vl + mrow;
#pragma unroll
    for (int i = 0; i < 8; ++i) {
      const int key = kt * 16 + i + half * 8;
      vrow[key * TT +  0] = acc0[i];
      vrow[key * TT + 16] = acc1[i];
      vrow[key * TT + 32] = acc2[i];
      vrow[key * TT + 48] = acc3[i];
    }
  }
#else
  // Fallback paths: 24 (key-tile x time-tile) pieces round-robined over waves.
  for (int tile = wave; tile < 24; tile += 8) {
    const int kt  = tile % 6;
    const int nt  = tile / 6;
    const int bk  = kb[kt * 16 + mrow];
    const float* Bcol = Ml + nt * 16 + mrow;
    float dfrag[8];
#if HAVE_WMMA_F16
    v8f acc = {0.f, 0.f, 0.f, 0.f, 0.f, 0.f, 0.f, 0.f};
    for (int k0 = 0; k0 < MEL_BINS; k0 += 32) {
      v16h a, bb;
#pragma unroll
      for (int i = 0; i < 8; ++i) {
        const int kA = k0 + (i < 4 ? 2 * i : 16 + 2 * (i - 4)) + half * 8;
        a[2 * i]     = (_Float16)a_entry(bk, kA);
        a[2 * i + 1] = (_Float16)a_entry(bk, kA + 1);
        const int kB = k0 + (i < 4 ? 2 * i : 16 + 2 * (i - 4)) + half * 8;
        bb[2 * i]     = (_Float16)Bcol[kB * TT];
        bb[2 * i + 1] = (_Float16)Bcol[(kB + 1) * TT];
      }
      acc = __builtin_amdgcn_wmma_f32_16x16x32_f16(
          false, a, false, bb, (short)0, acc, false, false);
    }
#pragma unroll
    for (int i = 0; i < 8; ++i) dfrag[i] = acc[i];
#else
    // Scalar reference path (host-compile pass / no-WMMA safety net).
    for (int i = 0; i < 8; ++i) dfrag[i] = 0.0f;
    for (int k0 = 0; k0 < MEL_BINS; ++k0)
      for (int i = 0; i < 8; ++i)
        dfrag[i] += a_entry(kb[kt * 16 + i + half * 8], k0) * Bcol[k0 * TT];
#endif
#pragma unroll
    for (int i = 0; i < 8; ++i)
      Vl[(kt * 16 + i + half * 8) * TT + nt * 16 + mrow] = dfrag[i];
  }
#endif
  __syncthreads();

  // ---- Rank-count selection: out = (#{j : v_j > v_k} <= 13) ---------------
  const int col  = tid & (TT - 1);   // time column
  const int part = tid >> 6;         // 0..3 -> keys [22*part, 22*part+22)
  float vals[NUM_KEYS];
#pragma unroll
  for (int j = 0; j < NUM_KEYS; ++j) vals[j] = Vl[j * TT + col];
  __syncthreads();                   // snapshot done before in-place overwrite

  for (int k = part * 22; k < part * 22 + 22; ++k) {
    const float vk = Vl[k * TT + col];     // only this thread touches this slot
    int c = 0;
#pragma unroll
    for (int j = 0; j < NUM_KEYS; ++j) c += (vals[j] > vk) ? 1 : 0;
    Vl[k * TT + col] = (c <= 13) ? 1.0f : 0.0f;
  }
  __syncthreads();

  // ---- Write probs tile twice (tuple output) ------------------------------
#if HAVE_TDM
  if (tid < 32) {
    float* out1 = out + ((size_t)b * NUM_KEYS) * T_TOTAL + t0;
    float* out2 = out1 + (size_t)NBATCH * NUM_KEYS * T_TOTAL;
    tdm_copy_2d(lds_off(Vl), out1, T_TOTAL, NUM_KEYS, TT, NUM_KEYS,
                (uint64_t)T_TOTAL, /*to_lds=*/false);
    tdm_copy_2d(lds_off(Vl), out2, T_TOTAL, NUM_KEYS, TT, NUM_KEYS,
                (uint64_t)T_TOTAL, /*to_lds=*/false);
    __builtin_amdgcn_s_wait_tensorcnt(0);
  }
#else
  for (int i = tid; i < NUM_KEYS * TT; i += NTHREADS) {
    const int k = i / TT, c = i % TT;
    const float p = Vl[k * TT + c];
    const size_t o = ((size_t)b * NUM_KEYS + k) * T_TOTAL + t0 + c;
    out[o] = p;
    out[o + (size_t)NBATCH * NUM_KEYS * T_TOTAL] = p;
  }
#endif
}

extern "C" void kernel_launch(void* const* d_in, const int* in_sizes, int n_in,
                              void* d_out, int out_size, void* d_ws, size_t ws_size,
                              hipStream_t stream) {
  (void)in_sizes; (void)n_in; (void)out_size; (void)d_ws; (void)ws_size;
  const float* mel      = (const float*)d_in[0];
  const int*   key_bins = (const int*)d_in[1];
  float*       out      = (float*)d_out;

  dim3 grid(T_TOTAL / TT, NBATCH);   // 128 time tiles x 32 batches
  dim3 block(NTHREADS);
  hps_key_quantile_kernel<<<grid, block, 0, stream>>>(mel, key_bins, out);
}
